// DeepSetArchitecture_85787676770956
// MI455X (gfx1250) — matrix operand order repaired and checked by reading, not compiled, not run on hardware
//
#include <hip/hip_runtime.h>

// ---------------------------------------------------------------------------
// DeepSet, compute-bound (~107 GFLOP vs <70MB HBM) -> bf16 WMMA, f32 accum.
// Transposed matmul (H^T = W^T * S^T): each wave owns 16 points; activations
// stay in registers across all layers (K-permutation baked into staged
// weights). A-operands (weights) stream from a 104KB LDS stage as
// conflict-free ds_load_b128. An asm memory clobber per chunk iteration pins
// those loads in-loop (prevents LICM register blow-up + scratch spills).
// Pack path: explicit paired v_cvt_pk_bf16_f32 + one v_pk_max_i16 per pair
// (integer-domain ReLU on the bf16 bit pattern; exact, -0 -> +0).
// ---------------------------------------------------------------------------

typedef __attribute__((ext_vector_type(16))) __bf16 v16bf;
typedef __attribute__((ext_vector_type(8)))  __bf16 v8bf;
typedef __attribute__((ext_vector_type(2)))  __bf16 v2bf;
typedef __attribute__((ext_vector_type(8)))  float  v8f;
typedef __attribute__((ext_vector_type(4)))  float  v4f;
typedef __attribute__((ext_vector_type(4)))  int    v4i;
typedef __attribute__((ext_vector_type(2)))  short  v2s;

union BF16x16 { v16bf v; v8bf h[2]; unsigned u[8]; };

#define BATCH 512
#define NPTS  2048
#define FEAT  16
#define PADV  (-10000.0f)

// workspace layout (bytes)
#define WS_POOL_BYTES ((size_t)BATCH * 128 * 4)          // pooled [512][128] f32
#define WS_PHI        (WS_POOL_BYTES)
#define PHI_LOFF(l)   (8192 + ((l) - 1) * 32768)         // l = 1..3 (layer0 at 0)
#define PHI_BYTES     (8192 + 3 * 32768)                 // 106496
#define WS_RHO        (WS_PHI + PHI_BYTES)
#define RHO_LOFF(l)   ((l) * 32768)                      // l = 0..2
#define RHO_BYTES     (3 * 32768)

__device__ __forceinline__ v8f wmma_bf16(v16bf a, v16bf b, v8f c) {
  return __builtin_amdgcn_wmma_f32_16x16x32_bf16(false, a, false, b, (short)0, c,
                                                 false, false);
}

__device__ __forceinline__ float relu(float x) {
  return fmaxf(x, 0.0f);
}

// (bf16(a), bf16(b)) packed into one u32 via a single v_cvt_pk_bf16_f32.
__device__ __forceinline__ unsigned cvt_pk_u32(float a, float b) {
#if __has_builtin(__builtin_amdgcn_cvt_pk_bf16_f32)
  v2bf p = __builtin_amdgcn_cvt_pk_bf16_f32(a, b);
  return __builtin_bit_cast(unsigned, p);
#else
  union { v2bf p; unsigned u; } t;
  t.p[0] = (__bf16)a; t.p[1] = (__bf16)b;
  return t.u;
#endif
}

// Packed bf16 ReLU on a (bf16,bf16) pair, integer domain: zero any half with
// the sign bit set. Clang folds this to a single v_pk_max_i16 (round-6 asm).
__device__ __forceinline__ unsigned relu_pk_u32(unsigned u) {
  v2s s = __builtin_bit_cast(v2s, u);
  s = s & ~(s >> 15);
  return __builtin_bit_cast(unsigned, s);
}

// A-fragment (weights) from LDS stage: lane's 32 contiguous bytes.
__device__ __forceinline__ v16bf ldA(const char* p) {
  BF16x16 t;
  t.h[0] = *(const v8bf*)p;          // ds_load_b128
  t.h[1] = *(const v8bf*)(p + 16);   // ds_load_b128
  return t.v;
}

// Fuse bias + paired bf16-pack + packed-integer ReLU of two accumulator tiles
// (nt0, nt0+1) into the next layer's B fragment (pure per-lane op thanks to
// the staged K-permutation). ReLU after RNE conversion is exact.
template <bool BIAS>
__device__ __forceinline__ v16bf packpair(v8f a0, v8f a1,
                                          const float* __restrict__ bias,
                                          int nt0, int H8) {
  BF16x16 o;
  if (BIAS) {
    v4f b00 = *(const v4f*)(bias + 16 * nt0 + H8);
    v4f b01 = *(const v4f*)(bias + 16 * nt0 + H8 + 4);
    v4f b10 = *(const v4f*)(bias + 16 * (nt0 + 1) + H8);
    v4f b11 = *(const v4f*)(bias + 16 * (nt0 + 1) + H8 + 4);
#pragma unroll
    for (int s = 0; s < 4; ++s) {
      const int e = 2 * s, f = 2 * s + 1;
      float e0 = a0[e] + (e < 4 ? b00[e] : b01[e - 4]);
      float e1 = a0[f] + (f < 4 ? b00[f] : b01[f - 4]);
      float f0 = a1[e] + (e < 4 ? b10[e] : b11[e - 4]);
      float f1 = a1[f] + (f < 4 ? b10[f] : b11[f - 4]);
      o.u[s]     = relu_pk_u32(cvt_pk_u32(e0, e1));
      o.u[4 + s] = relu_pk_u32(cvt_pk_u32(f0, f1));
    }
  } else {  // bias already baked into the matmul (layer 0 ones-row trick)
#pragma unroll
    for (int s = 0; s < 4; ++s) {
      o.u[s]     = relu_pk_u32(cvt_pk_u32(a0[2 * s], a0[2 * s + 1]));
      o.u[4 + s] = relu_pk_u32(cvt_pk_u32(a1[2 * s], a1[2 * s + 1]));
    }
  }
  return o.v;
}

// ---------------------------------------------------------------------------
// Prep: fragment all weight matrices into bf16 A-operand layout in workspace.
// Stage contract: frag(nt,kc), lane L (H=L>>4): h[i] = Weff[kc*32+(i<8 ? 8H+i : 16+8H+i-8)][16nt+(L&15)].
// phi layer0: Weff rows 0..15 = w0, row 16 = b0 (ones-row bias), 17..31 = 0.
// ---------------------------------------------------------------------------
__global__ void prep_weights(const float* __restrict__ w0, const float* __restrict__ b0,
                             const float* __restrict__ w1, const float* __restrict__ w2,
                             const float* __restrict__ w3, const float* __restrict__ r0,
                             const float* __restrict__ r1, const float* __restrict__ r2,
                             char* __restrict__ ws) {
  const int t = blockIdx.x * blockDim.x + threadIdx.x;
  if (t >= 200 * 32) return;
  const int frag = t >> 5, L = t & 31, H = L >> 4;
  const float* W;
  const float* brow = nullptr;
  int kbase = 0, kmax = 128, nt;
  size_t out;
  if (frag < 8) {                       // phi layer 0 (K=16 padded to 32)
    W = w0; brow = b0; kmax = 16; nt = frag;
    out = WS_PHI + (size_t)frag * 1024;
  } else if (frag < 104) {              // phi layers 1..3
    int idx = frag - 8, l = idx >> 5, r = idx & 31;
    W = (l == 0) ? w1 : (l == 1) ? w2 : w3;
    nt = r >> 2; kbase = (r & 3) * 32;
    out = WS_PHI + PHI_LOFF(l + 1) + (size_t)r * 1024;
  } else {                              // rho layers 0..2
    int idx = frag - 104, l = idx >> 5, r = idx & 31;
    W = (l == 0) ? r0 : (l == 1) ? r1 : r2;
    nt = r >> 2; kbase = (r & 3) * 32;
    out = WS_RHO + RHO_LOFF(l) + (size_t)r * 1024;
  }
  const int n = 16 * nt + (L & 15);
  BF16x16 o;
#pragma unroll
  for (int i = 0; i < 16; ++i) {
    int k = kbase + (i < 8 ? 8 * H + i : 16 + 8 * H + (i - 8));
    float v = (k < kmax) ? W[(size_t)k * 128 + n]
                         : ((brow != nullptr && k == kmax) ? brow[n] : 0.0f);
    o.v[i] = (__bf16)v;
  }
  *(v16bf*)(ws + out + (size_t)L * 32) = o.v;
}

// ---------------------------------------------------------------------------
// Kernel 1: phi + masked pooling. 512 blocks (one per b) x 8 waves; each wave
// handles 16 points per 128-point chunk. No barriers inside the chunk loop.
// ---------------------------------------------------------------------------
__global__ __launch_bounds__(256) void deepset_phi_pool(
    const float* __restrict__ x,
    const float* __restrict__ b1, const float* __restrict__ b2,
    const float* __restrict__ b3,
    const char* __restrict__ wsStage,   // ws + WS_PHI
    float* __restrict__ pooled) {
  extern __shared__ __align__(16) char smem[];   // PHI_BYTES weight stage
  __shared__ float pool_lds[128];
  __shared__ int s_len;

  const int tid = threadIdx.x, lane = tid & 31, w = tid >> 5;
  const int j = lane & 15, H = lane >> 4, H8 = H * 8;
  const size_t laneB = (size_t)lane * 32;
  const int bidx = blockIdx.x;

  // one-time: copy pre-fragmented weights into LDS
  for (int o = tid * 16; o < (int)PHI_BYTES; o += 256 * 16)
    *(v4i*)(smem + o) = *(const v4i*)(wsStage + o);

  if (tid == 0) s_len = NPTS;
  if (tid < 128) pool_lds[tid] = 0.0f;
  __syncthreads();

  // prefix-mask length: first all-PAD row
  const float* xb = x + (size_t)bidx * NPTS * FEAT;
  int myMin = NPTS;
  for (int n = tid; n < NPTS; n += 256) {
    const v4f* r = (const v4f*)(xb + (size_t)n * FEAT);
    v4f a0 = r[0], a1 = r[1], a2 = r[2], a3 = r[3];
    bool pad = true;
#pragma unroll
    for (int i = 0; i < 4; ++i)
      pad = pad && (a0[i] == PADV) && (a1[i] == PADV) && (a2[i] == PADV) && (a3[i] == PADV);
    if (pad && n < myMin) myMin = n;
  }
  atomicMin(&s_len, myMin);
  __syncthreads();
  const int len = s_len;
  const int nchunks = (len + 127) >> 7;

  v8f pool[8];
  {
    v8f z = {};
#pragma unroll
    for (int nt = 0; nt < 8; ++nt) pool[nt] = z;
  }

  const int mloc = w * 16 + j;  // point index within chunk (same for both halves)
  for (int ch = 0; ch < nchunks; ++ch) {
    // Pin the (loop-invariant) LDS weight-fragment loads inside the loop:
    // without this, LICM hoists ~100 fragments, exceeds 256 VGPRs and spills.
    asm volatile("" ::: "memory");

    const int m = ch * 128 + mloc;
    const bool valid = m < len;

    // ---- B'x straight from global: lane-lo feats 0..7, lane-hi feats 8..15;
    //      slot v=4 of lane-lo carries the bias ones-row (k=16). ----
    const float* xr = xb + (size_t)m * FEAT + H8;
    v4f x0 = *(const v4f*)xr;
    v4f x1 = *(const v4f*)(xr + 4);
    BF16x16 bx;
    bx.u[0] = cvt_pk_u32(x0[0], x0[1]);
    bx.u[1] = cvt_pk_u32(x0[2], x0[3]);
    bx.u[2] = cvt_pk_u32(x1[0], x1[1]);
    bx.u[3] = cvt_pk_u32(x1[2], x1[3]);
    bx.u[4] = H ? 0u : 0x00003F80u;   // (1.0bf16, 0) ones-row for bias
    bx.u[5] = 0u; bx.u[6] = 0u; bx.u[7] = 0u;

    if (ch + 1 < nchunks)
      __builtin_prefetch(xb + (size_t)(m + 128) * FEAT, 0, 1);

    v16bf Bc[4], Bn[4];

    // ---- layer 0 (K=32, one frag per nt; bias baked in) ----
#pragma unroll
    for (int kco = 0; kco < 4; ++kco) {
      v8f a0 = {}, a1 = {};
      a0 = wmma_bf16(ldA(smem + (size_t)(2 * kco) * 1024 + laneB), bx.v, a0);
      a1 = wmma_bf16(ldA(smem + (size_t)(2 * kco + 1) * 1024 + laneB), bx.v, a1);
      Bc[kco] = packpair<false>(a0, a1, nullptr, 0, H8);
    }

    // ---- layer 1 ----
#pragma unroll
    for (int kco = 0; kco < 4; ++kco) {
      v8f a0 = {}, a1 = {};
#pragma unroll
      for (int kc = 0; kc < 4; ++kc)
        a0 = wmma_bf16(ldA(smem + PHI_LOFF(1) + (size_t)((2 * kco) * 4 + kc) * 1024 + laneB), Bc[kc], a0);
#pragma unroll
      for (int kc = 0; kc < 4; ++kc)
        a1 = wmma_bf16(ldA(smem + PHI_LOFF(1) + (size_t)((2 * kco + 1) * 4 + kc) * 1024 + laneB), Bc[kc], a1);
      Bn[kco] = packpair<true>(a0, a1, b1, 2 * kco, H8);
    }

    // ---- layer 2 ----
#pragma unroll
    for (int kco = 0; kco < 4; ++kco) {
      v8f a0 = {}, a1 = {};
#pragma unroll
      for (int kc = 0; kc < 4; ++kc)
        a0 = wmma_bf16(ldA(smem + PHI_LOFF(2) + (size_t)((2 * kco) * 4 + kc) * 1024 + laneB), Bn[kc], a0);
#pragma unroll
      for (int kc = 0; kc < 4; ++kc)
        a1 = wmma_bf16(ldA(smem + PHI_LOFF(2) + (size_t)((2 * kco + 1) * 4 + kc) * 1024 + laneB), Bn[kc], a1);
      Bc[kco] = packpair<true>(a0, a1, b2, 2 * kco, H8);
    }

    // ---- layer 3 + masked pooling (register accumulation) ----
#pragma unroll
    for (int nt = 0; nt < 8; ++nt) {
      v8f a = {};
#pragma unroll
      for (int kc = 0; kc < 4; ++kc)
        a = wmma_bf16(ldA(smem + PHI_LOFF(3) + (size_t)(nt * 4 + kc) * 1024 + laneB), Bc[kc], a);
      if (valid) {
        v4f bb0 = *(const v4f*)(b3 + 16 * nt + H8);
        v4f bb1 = *(const v4f*)(b3 + 16 * nt + H8 + 4);
#pragma unroll
        for (int r = 0; r < 8; ++r)
          pool[nt][r] += relu(a[r] + (r < 4 ? bb0[r] : bb1[r - 4]));
      }
    }
  }

  // one-time reduction: per-lane partials -> pooled[b][:]
#pragma unroll
  for (int nt = 0; nt < 8; ++nt)
#pragma unroll
    for (int r = 0; r < 8; ++r)
      atomicAdd(&pool_lds[16 * nt + H8 + r], pool[nt][r]);   // ds_add_f32
  __syncthreads();
  if (tid < 128) pooled[(size_t)bidx * 128 + tid] = pool_lds[tid];
}

// ---------------------------------------------------------------------------
// Kernel 2: rho. 4 blocks x 8 waves x 16 points. Same register-resident chain.
// ---------------------------------------------------------------------------
__global__ __launch_bounds__(256) void deepset_rho(
    const float* __restrict__ pooled,
    const float* __restrict__ rb0, const float* __restrict__ rb1,
    const float* __restrict__ rb2,
    const float* __restrict__ w3, const float* __restrict__ b3,
    const char* __restrict__ wsStage,   // ws + WS_RHO
    float* __restrict__ out) {
  extern __shared__ __align__(16) char smem[];   // RHO_BYTES weight stage

  const int tid = threadIdx.x, lane = tid & 31, w = tid >> 5;
  const int j = lane & 15, H = lane >> 4, H8 = H * 8;
  const size_t laneB = (size_t)lane * 32;
  const int m = blockIdx.x * 128 + w * 16 + j;

  for (int o = tid * 16; o < (int)RHO_BYTES; o += 256 * 16)
    *(v4i*)(smem + o) = *(const v4i*)(wsStage + o);
  __syncthreads();

  // B' from pooled memory, in the stage-consistent K order:
  // lane-lo: k {0..7},{16..23}; lane-hi: k {8..15},{24..31} per kc.
  v16bf Bc[4], Bn[4];
  const float* pr = pooled + (size_t)m * 128;
#pragma unroll
  for (int kc = 0; kc < 4; ++kc) {
    const float* p0 = pr + kc * 32 + H8;
    const float* p1 = pr + kc * 32 + 16 + H8;
    v4f c0 = *(const v4f*)p0, c1 = *(const v4f*)(p0 + 4);
    v4f c2 = *(const v4f*)p1, c3 = *(const v4f*)(p1 + 4);
    BF16x16 o;
    o.u[0] = cvt_pk_u32(c0[0], c0[1]); o.u[1] = cvt_pk_u32(c0[2], c0[3]);
    o.u[2] = cvt_pk_u32(c1[0], c1[1]); o.u[3] = cvt_pk_u32(c1[2], c1[3]);
    o.u[4] = cvt_pk_u32(c2[0], c2[1]); o.u[5] = cvt_pk_u32(c2[2], c2[3]);
    o.u[6] = cvt_pk_u32(c3[0], c3[1]); o.u[7] = cvt_pk_u32(c3[2], c3[3]);
    Bc[kc] = o.v;
  }

  // rho layer 0
#pragma unroll
  for (int kco = 0; kco < 4; ++kco) {
    v8f a0 = {}, a1 = {};
#pragma unroll
    for (int kc = 0; kc < 4; ++kc)
      a0 = wmma_bf16(ldA(smem + RHO_LOFF(0) + (size_t)((2 * kco) * 4 + kc) * 1024 + laneB), Bc[kc], a0);
#pragma unroll
    for (int kc = 0; kc < 4; ++kc)
      a1 = wmma_bf16(ldA(smem + RHO_LOFF(0) + (size_t)((2 * kco + 1) * 4 + kc) * 1024 + laneB), Bc[kc], a1);
    Bn[kco] = packpair<true>(a0, a1, rb0, 2 * kco, H8);
  }
  // rho layer 1
#pragma unroll
  for (int kco = 0; kco < 4; ++kco) {
    v8f a0 = {}, a1 = {};
#pragma unroll
    for (int kc = 0; kc < 4; ++kc)
      a0 = wmma_bf16(ldA(smem + RHO_LOFF(1) + (size_t)((2 * kco) * 4 + kc) * 1024 + laneB), Bn[kc], a0);
#pragma unroll
    for (int kc = 0; kc < 4; ++kc)
      a1 = wmma_bf16(ldA(smem + RHO_LOFF(1) + (size_t)((2 * kco + 1) * 4 + kc) * 1024 + laneB), Bn[kc], a1);
    Bc[kco] = packpair<true>(a0, a1, rb1, 2 * kco, H8);
  }

  // rho layer 2 fused with final 128->1 projection
  float partial = 0.0f;
#pragma unroll
  for (int nt = 0; nt < 8; ++nt) {
    v8f a = {};
#pragma unroll
    for (int kc = 0; kc < 4; ++kc)
      a = wmma_bf16(ldA(smem + RHO_LOFF(2) + (size_t)(nt * 4 + kc) * 1024 + laneB), Bc[kc], a);
    v4f bb0 = *(const v4f*)(rb2 + 16 * nt + H8);
    v4f bb1 = *(const v4f*)(rb2 + 16 * nt + H8 + 4);
    v4f w30 = *(const v4f*)(w3 + 16 * nt + H8);
    v4f w31 = *(const v4f*)(w3 + 16 * nt + H8 + 4);
#pragma unroll
    for (int r = 0; r < 8; ++r) {
      float t = relu(a[r] + (r < 4 ? bb0[r] : bb1[r - 4]));
      partial += t * (r < 4 ? w30[r] : w31[r - 4]);
    }
  }
  // combine the two lane halves (same m, complementary n sets): SWAPX16 swizzle
  int other = __builtin_amdgcn_ds_swizzle(__float_as_int(partial), 0x401F);
  float sum = partial + __int_as_float(other);
  if (H == 0) out[m] = sum + b3[0];
}

// ---------------------------------------------------------------------------
extern "C" void kernel_launch(void* const* d_in, const int* in_sizes, int n_in,
                              void* d_out, int out_size, void* d_ws, size_t ws_size,
                              hipStream_t stream) {
  const float* x      = (const float*)d_in[0];
  const float* phi_w0 = (const float*)d_in[1];
  const float* phi_b0 = (const float*)d_in[2];
  const float* phi_w1 = (const float*)d_in[3];
  const float* phi_b1 = (const float*)d_in[4];
  const float* phi_w2 = (const float*)d_in[5];
  const float* phi_b2 = (const float*)d_in[6];
  const float* phi_w3 = (const float*)d_in[7];
  const float* phi_b3 = (const float*)d_in[8];
  const float* rho_w0 = (const float*)d_in[9];
  const float* rho_b0 = (const float*)d_in[10];
  const float* rho_w1 = (const float*)d_in[11];
  const float* rho_b1 = (const float*)d_in[12];
  const float* rho_w2 = (const float*)d_in[13];
  const float* rho_b2 = (const float*)d_in[14];
  const float* rho_w3 = (const float*)d_in[15];
  const float* rho_b3 = (const float*)d_in[16];

  char*  ws     = (char*)d_ws;
  float* pooled = (float*)d_ws;   // [512][128] f32 at offset 0

  prep_weights<<<25, 256, 0, stream>>>(phi_w0, phi_b0, phi_w1, phi_w2, phi_w3,
                                       rho_w0, rho_w1, rho_w2, ws);

  deepset_phi_pool<<<BATCH, 256, PHI_BYTES, stream>>>(
      x, phi_b1, phi_b2, phi_b3, ws + WS_PHI, pooled);

  deepset_rho<<<BATCH / 128, 256, RHO_BYTES, stream>>>(
      pooled, rho_b0, rho_b1, rho_b2, rho_w3, rho_b3, ws + WS_RHO,
      (float*)d_out);
}
